// Block_83116207112284
// MI455X (gfx1250) — matrix operand order repaired and checked by reading, not compiled, not run on hardware
//
#include <hip/hip_runtime.h>
#include <hip/hip_bf16.h>

// ---------------------------------------------------------------------------
// Shapes (fixed by the reference):
//   x: (32, 512, 128) -> X: (4096, 512);  lin1_w: (512,512); lin2_w: (256,512)
//   H1 = X @ W1^T : (4096, 512);  H = H1 @ W2^T : (4096, 256)
//   out = H + mamba1_scan_per_row(H)   (FFT / mamba2 branch is identically ~0)
// ---------------------------------------------------------------------------

typedef __attribute__((ext_vector_type(16))) _Float16 v16h;
typedef __attribute__((ext_vector_type(8)))  _Float16 v8h;
typedef __attribute__((ext_vector_type(8)))  float    v8f;
typedef int v4i __attribute__((vector_size(4 * sizeof(int))));

#define BH    32
#define LSEQ  512
#define CCH   128
#define N1    512
#define N2    256
#define RTOT  4096   // B*C rows

// ------------------- CDNA5 async global->LDS helpers -----------------------

#if __has_builtin(__builtin_amdgcn_global_load_async_to_lds_b128)
#define HAVE_ASYNC_LDS 1
__device__ __forceinline__ void async_cp_b128(const _Float16* g, _Float16* l) {
  __builtin_amdgcn_global_load_async_to_lds_b128(
      (__attribute__((address_space(1))) v4i*)g,
      (__attribute__((address_space(3))) v4i*)l, 0, 0);
}
#else
#define HAVE_ASYNC_LDS 0
__device__ __forceinline__ void async_cp_b128(const _Float16* g, _Float16* l) {
  *(v8h*)l = *(const v8h*)g;   // synchronous fallback
}
#endif

template <int N>
__device__ __forceinline__ void wait_async() {
#if HAVE_ASYNC_LDS
#if __has_builtin(__builtin_amdgcn_s_wait_asynccnt)
  __builtin_amdgcn_s_wait_asynccnt(N);
#else
  if (N == 0) asm volatile("s_wait_asynccnt 0x0" ::: "memory");
  else        asm volatile("s_wait_asynccnt 0x1" ::: "memory");
#endif
#endif
}

__device__ __forceinline__ v8f wmma16(v16h a, v16h b, v8f c) {
  // (neg_a, A, neg_b, B, c_mod, C, reuse_a, reuse_b)
  return __builtin_amdgcn_wmma_f32_16x16x32_f16(false, a, false, b,
                                                (short)0, c, false, false);
}

// ----------------------------- converters ----------------------------------

__global__ __launch_bounds__(256) void k_cvt_f16(const float* __restrict__ in,
                                                 _Float16* __restrict__ out, int n) {
  int i = blockIdx.x * blockDim.x + threadIdx.x;
  if (i < n) out[i] = (_Float16)in[i];
}

// Tiled transpose-convert: X[(b*C + c)*L + l] = (f16) x[b*L*C + l*C + c]
// Coalesced on both sides via a 32x33 LDS tile.
__global__ __launch_bounds__(256) void k_pack_x(const float* __restrict__ x,
                                                _Float16* __restrict__ X) {
  __shared__ float tile[32][33];
  const int b  = blockIdx.z;
  const int l0 = blockIdx.y * 32;
  const int c0 = blockIdx.x * 32;
  const int tx = threadIdx.x;          // 0..31
  const int ty = threadIdx.y;          // 0..7
#pragma unroll
  for (int k = 0; k < 4; ++k)
    tile[ty + 8 * k][tx] = x[((size_t)b * LSEQ + l0 + ty + 8 * k) * CCH + c0 + tx];
  __syncthreads();
#pragma unroll
  for (int k = 0; k < 4; ++k)
    X[((size_t)b * CCH + c0 + ty + 8 * k) * LSEQ + l0 + tx] =
        (_Float16)tile[tx][ty + 8 * k];
}

// ----------------------------- WMMA GEMM -----------------------------------
// C(M x N) = A(M x K, f16 row-major) * Bw(N x K, f16 row-major)^T + bias[n]
// Block = 4 waves; block tile 128(M) x 32(N); each wave owns 32x32 (2x2
// accumulators). B tile (32x32 f16 = 2 KB) is shared by all waves and staged
// through LDS with double-buffered GLOBAL_LOAD_ASYNC_TO_LDS_B128 (ASYNCcnt).
// A fragments load straight from global (already WMMA-layout contiguous).
//
// WMMA 16-bit A 16x32: lane<16 -> K {k0..k0+7, k0+16..k0+23}, M = lane&15
// WMMA 16-bit B 32x16: lane<16 -> K {k0..k0+7, k0+16..k0+23}, N = lane&15; hi half K+8
// D: lane 0-15 -> M = v, lane 16-31 -> M = v+8; N = lane&15
__global__ __launch_bounds__(128) void k_wmma_gemm(
    const _Float16* __restrict__ A, const _Float16* __restrict__ Bw,
    const float* __restrict__ bias, int N, int K,
    _Float16* __restrict__ Ch,       // optional f16 row-major output (M x N)
    float* __restrict__ Ct, int ldt) // optional f32 transposed output (N x ldt)
{
  __shared__ _Float16 Bsh[2][32 * 32];   // double-buffered B tile

  const int lane = threadIdx.x;
  const int wid  = threadIdx.y;          // 0..3
  const int tid  = wid * 32 + lane;      // 0..127
  const int half = lane >> 4;
  const int lid  = lane & 15;
  const int n0 = blockIdx.x * 32;
  const int m0 = blockIdx.y * 128 + wid * 32;

  // B staging: thread -> (column n, 8-half K chunk); 128 threads cover 32x32.
  const int bn = tid >> 2;
  const int bk = (tid & 3) * 8;
  const _Float16* bsrc = Bw + (size_t)(n0 + bn) * K + bk;
  const int bdst = bn * 32 + bk;

  const _Float16* arow0 = A + (size_t)(m0 + lid) * K;
  const _Float16* arow1 = arow0 + (size_t)16 * K;

  v8f acc00 = {}, acc01 = {}, acc10 = {}, acc11 = {};

  async_cp_b128(bsrc, &Bsh[0][bdst]);          // prologue: tile for k0 = 0

  const int nit = K / 32;
  for (int it = 0; it < nit; ++it) {
    const int k0 = it * 32;
    if (it + 1 < nit) {
      async_cp_b128(bsrc + k0 + 32, &Bsh[(it + 1) & 1][bdst]);
      wait_async<1>();                         // current tile done, next in flight
    } else {
      wait_async<0>();
    }
    __syncthreads();

    const _Float16* bbuf = &Bsh[it & 1][0];
    v16h a0, a1;
    const int ka = k0 + half * 8;
    ((v8h*)&a0)[0] = *(const v8h*)(arow0 + ka);
    ((v8h*)&a0)[1] = *(const v8h*)(arow0 + ka + 16);
    ((v8h*)&a1)[0] = *(const v8h*)(arow1 + ka);
    ((v8h*)&a1)[1] = *(const v8h*)(arow1 + ka + 16);
    const int kb = half * 8;
    v16h b0, b1;
    ((v8h*)&b0)[0] = *(const v8h*)(bbuf + lid * 32 + kb);
    ((v8h*)&b0)[1] = *(const v8h*)(bbuf + lid * 32 + kb + 16);
    ((v8h*)&b1)[0] = *(const v8h*)(bbuf + (16 + lid) * 32 + kb);
    ((v8h*)&b1)[1] = *(const v8h*)(bbuf + (16 + lid) * 32 + kb + 16);

    acc00 = wmma16(a0, b0, acc00);
    acc01 = wmma16(a0, b1, acc01);
    acc10 = wmma16(a1, b0, acc10);
    acc11 = wmma16(a1, b1, acc11);
    __syncthreads();
  }

  const float bn0 = bias[n0 + lid];
  const float bn1 = bias[n0 + 16 + lid];
#pragma unroll
  for (int v = 0; v < 8; ++v) {
    const int m_lo = m0 + half * 8 + v;
    const int m_hi = m_lo + 16;
    const float v00 = acc00[v] + bn0, v01 = acc01[v] + bn1;
    const float v10 = acc10[v] + bn0, v11 = acc11[v] + bn1;
    if (Ch) {
      Ch[(size_t)m_lo * N + n0 + lid]      = (_Float16)v00;
      Ch[(size_t)m_lo * N + n0 + 16 + lid] = (_Float16)v01;
      Ch[(size_t)m_hi * N + n0 + lid]      = (_Float16)v10;
      Ch[(size_t)m_hi * N + n0 + 16 + lid] = (_Float16)v11;
    }
    if (Ct) {
      Ct[(size_t)(n0 + lid) * ldt + m_lo]      = v00;
      Ct[(size_t)(n0 + 16 + lid) * ldt + m_lo] = v01;
      Ct[(size_t)(n0 + lid) * ldt + m_hi]      = v10;
      Ct[(size_t)(n0 + 16 + lid) * ldt + m_hi] = v11;
    }
  }
}

// ----------------------------- Mamba scan ----------------------------------

__device__ __forceinline__ float silu_f(float v) { return v / (1.f + __expf(-v)); }
__device__ __forceinline__ float softplus_f(float v) {
  return (v > 20.f) ? v : __logf(1.f + __expf(v));
}

// Ht: (N2 x RTOT) f32, transposed so reads are coalesced across lanes.
// One row per thread; d_model=1, d_inner=2, d_state=16, d_conv=4.
__global__ __launch_bounds__(256) void k_mamba_scan(
    const float* __restrict__ Ht,
    const float* __restrict__ in_w,   const float* __restrict__ conv_w,
    const float* __restrict__ conv_b, const float* __restrict__ xproj_w,
    const float* __restrict__ dt_w,   const float* __restrict__ dt_b,
    const float* __restrict__ A_log,  const float* __restrict__ Dp,
    const float* __restrict__ out_w,  float* __restrict__ out)
{
  const int r = blockIdx.x * blockDim.x + threadIdx.x;

  const float s0 = in_w[0], s1 = in_w[1], z0w = in_w[2], z1w = in_w[3];
  const float cw00 = conv_w[0], cw01 = conv_w[1], cw02 = conv_w[2], cw03 = conv_w[3];
  const float cw10 = conv_w[4], cw11 = conv_w[5], cw12 = conv_w[6], cw13 = conv_w[7];
  const float cb0 = conv_b[0], cb1 = conv_b[1];

  float xpw[66];                        // xproj_w (33 x 2) row-major
#pragma unroll
  for (int i = 0; i < 66; ++i) xpw[i] = xproj_w[i];

  const float dtw0 = dt_w[0], dtw1 = dt_w[1];
  const float dtb0 = dt_b[0], dtb1 = dt_b[1];

  float A0[16], A1[16];
#pragma unroll
  for (int nn = 0; nn < 16; ++nn) {
    A0[nn] = -__expf(A_log[nn]);
    A1[nn] = -__expf(A_log[16 + nn]);
  }
  const float D0 = Dp[0], D1 = Dp[1];
  const float ow0 = out_w[0], ow1 = out_w[1];

  float st0[16], st1[16];
#pragma unroll
  for (int nn = 0; nn < 16; ++nn) { st0[nn] = 0.f; st1[nn] = 0.f; }

  float um3 = 0.f, um2 = 0.f, um1 = 0.f;   // causal conv window (left pad 3)

  for (int l = 0; l < N2; ++l) {
    const float u = Ht[(size_t)l * RTOT + r];

    // depthwise causal conv (xc[l][d] = u*in_w[d] folded through linearity)
    const float a0 = um3 * cw00 + um2 * cw01 + um1 * cw02 + u * cw03;
    const float a1 = um3 * cw10 + um2 * cw11 + um1 * cw12 + u * cw13;
    const float xs0 = silu_f(s0 * a0 + cb0);
    const float xs1 = silu_f(s1 * a1 + cb1);
    um3 = um2; um2 = um1; um1 = u;

    const float dtr = xs0 * xpw[0] + xs1 * xpw[1];
    const float dt0 = softplus_f(dtr * dtw0 + dtb0);
    const float dt1 = softplus_f(dtr * dtw1 + dtb1);
    const float g0 = silu_f(z0w * u), g1 = silu_f(z1w * u);

    float y0 = xs0 * D0, y1 = xs1 * D1;
    const float db0 = dt0 * xs0, db1 = dt1 * xs1;
#pragma unroll
    for (int nn = 0; nn < 16; ++nn) {
      const float Bn = xs0 * xpw[2 + 2 * nn]  + xs1 * xpw[3 + 2 * nn];
      const float Cn = xs0 * xpw[34 + 2 * nn] + xs1 * xpw[35 + 2 * nn];
      st0[nn] = __expf(dt0 * A0[nn]) * st0[nn] + db0 * Bn;
      st1[nn] = __expf(dt1 * A1[nn]) * st1[nn] + db1 * Bn;
      y0 += st0[nn] * Cn;
      y1 += st1[nn] * Cn;
    }
    const float mres = (y0 * g0) * ow0 + (y1 * g1) * ow1;
    out[(size_t)r * N2 + l] = u + mres;   // residual x_res fused in
  }
}

// ----------------------------- launcher ------------------------------------

extern "C" void kernel_launch(void* const* d_in, const int* in_sizes, int n_in,
                              void* d_out, int out_size, void* d_ws, size_t ws_size,
                              hipStream_t stream) {
  (void)in_sizes; (void)n_in; (void)out_size; (void)ws_size;

  const float* x        = (const float*)d_in[0];
  const float* lin1_w   = (const float*)d_in[1];
  const float* lin1_b   = (const float*)d_in[2];
  const float* lin2_w   = (const float*)d_in[3];
  const float* lin2_b   = (const float*)d_in[4];
  const float* m1_in_w    = (const float*)d_in[5];
  const float* m1_conv_w  = (const float*)d_in[6];
  const float* m1_conv_b  = (const float*)d_in[7];
  const float* m1_xproj_w = (const float*)d_in[8];
  const float* m1_dt_w    = (const float*)d_in[9];
  const float* m1_dt_b    = (const float*)d_in[10];
  const float* m1_A_log   = (const float*)d_in[11];
  const float* m1_D       = (const float*)d_in[12];
  const float* m1_out_w   = (const float*)d_in[13];
  // d_in[14..22] (m2_*) unused: the series-decomp branch operates on a
  // length-1 axis, so res == 0 to fp32 rounding, st == 0, and mamba2's output
  // is O(eps^2) — identically negligible. Dropped.

  float* out = (float*)d_out;
  char* ws = (char*)d_ws;
  // workspace layout (needs ~13.4 MB):
  _Float16* Xh  = (_Float16*)(ws);                 // 4096*512*2  = 4,194,304
  _Float16* W1h = (_Float16*)(ws + 4194304);       //  512*512*2  =   524,288
  _Float16* W2h = (_Float16*)(ws + 4718592);       //  256*512*2  =   262,144
  _Float16* H1h = (_Float16*)(ws + 4980736);       // 4096*512*2  = 4,194,304
  float*    Htp = (float*)   (ws + 9175040);       //  256*4096*4 = 4,194,304

  k_pack_x <<<dim3(CCH / 32, LSEQ / 32, BH), dim3(32, 8), 0, stream>>>(x, Xh);
  k_cvt_f16<<<dim3((N1 * LSEQ + 255) / 256), dim3(256), 0, stream>>>(lin1_w, W1h, N1 * LSEQ);
  k_cvt_f16<<<dim3((N2 * N1 + 255) / 256),  dim3(256), 0, stream>>>(lin2_w, W2h, N2 * N1);

  // GEMM1: H1(4096x512,f16) = X @ W1^T + b1
  k_wmma_gemm<<<dim3(N1 / 32, RTOT / 128), dim3(32, 4), 0, stream>>>(
      Xh, W1h, lin1_b, N1, LSEQ, H1h, nullptr, 0);
  // GEMM2: Ht(256x4096,f32,transposed) = H1 @ W2^T + b2
  k_wmma_gemm<<<dim3(N2 / 32, RTOT / 128), dim3(32, 4), 0, stream>>>(
      H1h, W2h, lin2_b, N2, N1, nullptr, Htp, RTOT);

  k_mamba_scan<<<dim3(RTOT / 256), dim3(256), 0, stream>>>(
      Htp, m1_in_w, m1_conv_w, m1_conv_b, m1_xproj_w, m1_dt_w, m1_dt_b,
      m1_A_log, m1_D, m1_out_w, out);
}
